// Model_38414187495656
// MI455X (gfx1250) — matrix operand order repaired and checked by reading, not compiled, not run on hardware
//
#include <hip/hip_runtime.h>
#include <stdint.h>

// ---------------------------------------------------------------------------
// CDNA5 / gfx1250 types
// ---------------------------------------------------------------------------
typedef __attribute__((ext_vector_type(16))) __bf16   v16bf;
typedef __attribute__((ext_vector_type(8)))  float    v8f;
typedef __attribute__((ext_vector_type(4)))  uint32_t u32x4;
typedef __attribute__((ext_vector_type(4)))  float    f32x4;
typedef __attribute__((ext_vector_type(4)))  int      i32x4;

#define BS        32
#define LSEQ      1024
#define GSZ       1024
#define BAND_P    34     // half bandwidth of AtA
#define BAND_STR  36     // padded band storage stride

#if __has_builtin(__builtin_amdgcn_global_load_async_to_lds_b128) && \
    __has_builtin(__builtin_amdgcn_s_wait_asynccnt)
#define USE_ASYNC_LDS 1
typedef __attribute__((address_space(1))) i32x4* gas_i32x4p;
typedef __attribute__((address_space(3))) i32x4* las_i32x4p;
#else
#define USE_ASYNC_LDS 0
#endif

static __device__ __forceinline__ uint16_t f2bf(float f) {
    uint32_t u = __float_as_uint(f);
    u += 0x7FFFu + ((u >> 16) & 1u);        // round-to-nearest-even
    return (uint16_t)(u >> 16);
}

// 16-byte global -> LDS stage (async engine when available)
static __device__ __forceinline__ void stage16(const uint16_t* g, uint16_t* l) {
#if USE_ASYNC_LDS
    __builtin_amdgcn_global_load_async_to_lds_b128(
        (gas_i32x4p)(g), (las_i32x4p)(l), 0, 0);
#else
    *(u32x4*)l = *(const u32x4*)g;
#endif
}
static __device__ __forceinline__ void stage_wait() {
#if USE_ASYNC_LDS
    __builtin_amdgcn_s_wait_asynccnt(0);
#endif
    __syncthreads();
}

// ---------------------------------------------------------------------------
// Input transpose: u (32,100,1024) f32 -> X (32,1024,128) bf16, zero-padded
// ---------------------------------------------------------------------------
__global__ void k_transpose_in(const float* __restrict__ u,
                               uint16_t* __restrict__ X) {
    int c = threadIdx.x;                 // 0..127
    int l = blockIdx.x;                  // 0..1023
    int b = blockIdx.y;                  // 0..31
    float v = (c < 100) ? u[((size_t)b * 100 + c) * 1024 + l] : 0.f;
    X[((size_t)((b << 10) + l)) * 128 + c] = f2bf(v);
}

// ---------------------------------------------------------------------------
// Weight pack: w (O,C,5) f32 -> Wp (5,Opad,Cp) bf16, zero-padded
// ---------------------------------------------------------------------------
__global__ void k_pack_w(const float* __restrict__ w, uint16_t* __restrict__ dst,
                         int O, int C, int Opad, int Cp) {
    int c = threadIdx.x;
    int o = blockIdx.x;
    int k = blockIdx.y;
    float v = (o < O && c < C) ? w[((size_t)o * C + c) * 5 + k] : 0.f;
    dst[((size_t)(k * Opad + o)) * Cp + c] = f2bf(v);
}

// ---------------------------------------------------------------------------
// Conv layer as 5 accumulated WMMA GEMMs (one per tap, circular shift).
// Block = 64 o x 64 positions; 4 waves each own a 16-o subtile and share the
// position tile. The 68 needed X rows (64 + 2-halo each side) are staged into
// LDS once (async-to-LDS when available) and fragments come from ds_load_b128.
// LDS row stride = CP+8 halfs => stride % 32 dwords == 4, so the 16 fragment
// rows map to disjoint bank quads (minimal 2-pass b128 reads).
// A frag (16x32 bf16) per-lane layout (ISA 7.12.2):
//   lane m=lane&15, half=lane>>4; dword j: K = (j>>2)*16 + half*8 + (j&3)*2
// B frag (32x16 bf16): lane n=lane&15, half=lane>>4; dword j: K = (j>>2)*16 + half*8 + (j&3)*2
// ---------------------------------------------------------------------------
union FragBF { u32x4 q[2]; v16bf v; };

template <int CP, int OPAD, bool F32OUT, bool RELU>
__global__ __launch_bounds__(128)
void k_conv_wmma(const uint16_t* __restrict__ Xin, const uint16_t* __restrict__ Wp,
                 const float* __restrict__ bias, int Oreal,
                 void* __restrict__ out) {
    constexpr int KCH  = CP >> 5;              // K chunks of 32
    constexpr int ROWQ = CP >> 3;              // u32x4 per global bf16 row
    constexpr int LROW = CP + 8;               // LDS row stride in halfs
    constexpr int OSTR = F32OUT ? 128 : OPAD;  // output row stride (elements)

    __shared__ __align__(16) uint16_t sh[68 * LROW];

    const int lane = threadIdx.x & 31;
    const int wv   = threadIdx.x >> 5;
    const int half = lane >> 4;
    const int mn   = lane & 15;

    const int pos0 = blockIdx.x << 6;              // 64 positions per block
    const int b    = pos0 >> 10;
    const int l0   = pos0 & 1023;
    const int o0   = (blockIdx.y << 6) + (wv << 4);

    const u32x4* X4 = (const u32x4*)Xin;
    const u32x4* W4 = (const u32x4*)Wp;

    // ---- cooperative stage of 68 X rows into LDS --------------------------
    {
        const int total = 68 * ROWQ;               // 16B chunks
        for (int idx = threadIdx.x; idx < total; idx += 128) {
            const int r  = idx / ROWQ;
            const int qc = idx - r * ROWQ;
            const int lsrc = (l0 + r - 2) & 1023;  // circular halo
            stage16((const uint16_t*)(X4 + (size_t)((b << 10) + lsrc) * ROWQ + qc),
                    sh + (size_t)r * LROW + (qc << 3));
        }
        stage_wait();
    }

    v8f acc[4];
#pragma unroll
    for (int nt = 0; nt < 4; ++nt)
        acc[nt] = (v8f){0.f, 0.f, 0.f, 0.f, 0.f, 0.f, 0.f, 0.f};

#pragma unroll
    for (int k = 0; k < 5; ++k) {
        const u32x4* wrow = W4 + (size_t)(k * OPAD + (o0 + mn)) * ROWQ;
        if (k < 4)   // warm next tap's weight row (global_prefetch_b8)
            __builtin_prefetch((const void*)(wrow + (size_t)OPAD * ROWQ), 0, 1);
#pragma unroll
        for (int kc = 0; kc < KCH; ++kc) {
            FragBF a;
            a.q[0] = wrow[kc * 4 + half];          // dwords {0..3}+half*4
            a.q[1] = wrow[kc * 4 + 2 + half];      // dwords {8..11}+half*4
#pragma unroll
            for (int nt = 0; nt < 4; ++nt) {
                const int r = (nt << 4) + mn + k;  // staged row (halo offset)
                const u32x4* bq = (const u32x4*)(sh + (size_t)r * LROW);
                FragBF bf;
                bf.q[0] = bq[kc * 4 + half];       // ds_load_b128
                bf.q[1] = bq[kc * 4 + 2 + half];
                acc[nt] = __builtin_amdgcn_wmma_f32_16x16x32_bf16(
                    false, a.v, false, bf.v, (short)0, acc[nt], false, false);
            }
        }
    }

    // C/D layout: VGPR i holds (m = i + half*8, n = lane&15)
    const int moff = half << 3;
    float bv[8];
#pragma unroll
    for (int i = 0; i < 8; ++i) {
        int o = o0 + moff + i;
        bv[i] = (o < Oreal) ? bias[o] : 0.f;
    }
#pragma unroll
    for (int nt = 0; nt < 4; ++nt) {
        const int l = l0 + (nt << 4) + mn;
        float r[8];
#pragma unroll
        for (int i = 0; i < 8; ++i) {
            float v = acc[nt][i] + bv[i];
            if (RELU) v = fmaxf(v, 0.f);
            r[i] = v;
        }
        if (F32OUT) {
            float* dst = (float*)out + (size_t)((b << 10) + l) * OSTR + o0 + moff;
            f32x4 v0 = {r[0], r[1], r[2], r[3]};
            f32x4 v1 = {r[4], r[5], r[6], r[7]};
            *(f32x4*)(dst)     = v0;
            *(f32x4*)(dst + 4) = v1;
        } else {
            u32x4 pk;
#pragma unroll
            for (int i = 0; i < 4; ++i)
                pk[i] = (uint32_t)f2bf(r[2 * i]) |
                        ((uint32_t)f2bf(r[2 * i + 1]) << 16);
            uint16_t* dsth =
                (uint16_t*)out + (size_t)((b << 10) + l) * OSTR + o0 + moff;
            *(u32x4*)dsth = pk;
        }
    }
}

// ---------------------------------------------------------------------------
// param_net MLP: (1,64) -> 1024 -> 1024 -> 1024 -> 6   (single workgroup)
// ---------------------------------------------------------------------------
__global__ __launch_bounds__(1024)
void k_mlp(const float* __restrict__ pin,
           const float* __restrict__ w0, const float* __restrict__ b0,
           const float* __restrict__ w1, const float* __restrict__ b1,
           const float* __restrict__ w2, const float* __restrict__ b2,
           const float* __restrict__ w3, const float* __restrict__ b3,
           float* __restrict__ params) {
    __shared__ float sa[1024];
    __shared__ float sb[1024];
    int t = threadIdx.x;
    if (t < 64) sa[t] = pin[t];
    __syncthreads();
    float acc = b0[t];
    for (int i = 0; i < 64; ++i) acc += sa[i] * w0[i * 1024 + t];
    sb[t] = fmaxf(acc, 0.f);
    __syncthreads();
    acc = b1[t];
    for (int i = 0; i < 1024; ++i) acc += sb[i] * w1[i * 1024 + t];
    sa[t] = fmaxf(acc, 0.f);
    __syncthreads();
    acc = b2[t];
    for (int i = 0; i < 1024; ++i) acc += sa[i] * w2[i * 1024 + t];
    sb[t] = fmaxf(acc, 0.f);
    __syncthreads();
    if (t < 6) {
        acc = b3[t];
        for (int i = 0; i < 1024; ++i) acc += sb[i] * w3[i * 6 + t];
        params[t] = acc;
    }
}

// ---------------------------------------------------------------------------
// Step sizes: inv[0..30] = 1/clip(sigmoid(s0)), inv[32..62] = 1/clip(sigmoid(s1))
// ---------------------------------------------------------------------------
__global__ void k_steps(const float* __restrict__ s0l, const float* __restrict__ s1l,
                        float* __restrict__ inv) {
    int t = threadIdx.x;
    if (t < 31) {
        float s = 1.f / (1.f + __expf(-s0l[t]));
        s = fminf(fmaxf(s, 0.005f), 0.1f);
        inv[t] = 1.f / s;
    } else if (t >= 32 && t < 63) {
        float s = 1.f / (1.f + __expf(-s1l[t - 32]));
        s = fminf(fmaxf(s, 0.005f), 0.1f);
        inv[t] = 1.f / s;
    }
}

// ---------------------------------------------------------------------------
// up = mean over first 100 channels of last conv; p,q from params (2,3)
// ---------------------------------------------------------------------------
__global__ __launch_bounds__(1024)
void k_upq(const float* __restrict__ Y, const float* __restrict__ params,
           float* __restrict__ p, float* __restrict__ q) {
    int l = threadIdx.x, b = blockIdx.x;
    const float* row = Y + (size_t)((b << 10) + l) * 128;
    float s = 0.f;
    for (int o = 0; o < 100; ++o) s += row[o];
    float up = s * 0.01f;
    float up2 = up * up;
    p[(b << 10) + l] = params[0] + up * params[1] + up2 * params[2];
    q[(b << 10) + l] = params[3] + up * params[4] + up2 * params[5];
}

// ---------------------------------------------------------------------------
// Init band storage: ridge + BC one-hot rows contribute +1 on boundary diagonal
// ---------------------------------------------------------------------------
__global__ void k_init_band(float* __restrict__ band, float* __restrict__ Atb) {
    int idx = blockIdx.x * blockDim.x + threadIdx.x;
    const int total = BS * GSZ * BAND_STR;
    if (idx < total) {
        int d = idx % BAND_STR;
        int g = (idx / BAND_STR) % GSZ;
        float v = 0.f;
        if (d == 0) {
            int t = g >> 5, x = g & 31;
            v = 1e-6f + ((t == 0 || t == 31 || x == 0 || x == 31) ? 1.f : 0.f);
        }
        band[idx] = v;
    }
    if (idx < BS * GSZ) Atb[idx] = 0.f;
}

// ---------------------------------------------------------------------------
// Sparse AtA / Atb accumulation. Row (b,g) of A has <=7 nonzeros:
//   op1 = kron(D1t,I) (coef 1), op2 = kron(I,D1x)*p, op4 = kron(I,D1x^2)*q
// (rows 30,31 of D1x^2 are identically zero.)
// ---------------------------------------------------------------------------
__global__ void k_scatter(const float* __restrict__ p, const float* __restrict__ q,
                          const float* __restrict__ inv, const float* __restrict__ rhs,
                          float* __restrict__ band, float* __restrict__ Atb) {
    int idx = blockIdx.x * blockDim.x + threadIdx.x;   // b*1024+g
    if (idx >= BS * GSZ) return;
    int b = idx >> 10, g = idx & 1023;
    int t = g >> 5, x = g & 31;
    float pp = p[idx], qq = q[idx];
    const float* inv0 = inv;
    const float* inv1 = inv + 32;

    int   cols[7];
    float vals[7];
    // op1, coefficient 1.0
    float a = inv0[t < 31 ? t : 30];
    if (t < 31) { cols[0] = g;      vals[0] = -a; cols[1] = g + 32; vals[1] =  a; }
    else        { cols[0] = g - 32; vals[0] = -a; cols[1] = g;      vals[1] =  a; }
    // op2 * p
    float bb = inv1[x < 31 ? x : 30];
    if (x < 31) { cols[2] = g;     vals[2] = -bb * pp; cols[3] = g + 1; vals[3] = bb * pp; }
    else        { cols[2] = g - 1; vals[2] = -bb * pp; cols[3] = g;     vals[3] = bb * pp; }
    // op4 * q (zero for x >= 30)
    if (x < 30) {
        float c0 = inv1[x], c1 = inv1[x + 1];
        cols[4] = g;     vals[4] =  c0 * c0 * qq;
        cols[5] = g + 1; vals[5] = -c0 * (c0 + c1) * qq;
        cols[6] = g + 2; vals[6] =  c0 * c1 * qq;
    } else {
        cols[4] = g; vals[4] = 0.f;
        cols[5] = g; vals[5] = 0.f;
        cols[6] = g; vals[6] = 0.f;
    }

    // outer product row^T row into symmetric band (upper: band[ci][cj-ci])
#pragma unroll
    for (int i = 0; i < 7; ++i) {
#pragma unroll
        for (int j = 0; j < 7; ++j) {
            int ci = cols[i], cj = cols[j];
            if (ci <= cj) {
                atomicAdd(&band[((size_t)b * GSZ + ci) * BAND_STR + (cj - ci)],
                          vals[i] * vals[j]);
            }
        }
    }
    float r = rhs[idx];
#pragma unroll
    for (int i = 0; i < 7; ++i)
        atomicAdd(&Atb[(b << 10) + cols[i]], vals[i] * r);
}

// ---------------------------------------------------------------------------
// Banded Cholesky (half-bandwidth 34) + forward/back substitution, per batch.
// band[j][d] = M[j+d][j]; in-place factorization L, then solve.
// ---------------------------------------------------------------------------
__global__ __launch_bounds__(128)
void k_cholsolve(float* __restrict__ band, float* __restrict__ Atb,
                 float* __restrict__ out) {
    int b = blockIdx.x;
    float* B  = band + (size_t)b * GSZ * BAND_STR;
    float* rb = Atb + (b << 10);
    const int t = threadIdx.x;
    for (int j = 0; j < GSZ; ++j) {
        float* rowj = B + (size_t)j * BAND_STR;
        if (t == 0) rowj[0] = sqrtf(fmaxf(rowj[0], 1e-30f));
        __syncthreads();
        float invd = 1.f / rowj[0];
        int kmax = min(BAND_P, GSZ - 1 - j);
        for (int d = 1 + t; d <= kmax; d += 128) rowj[d] *= invd;
        __syncthreads();
        for (int k = 1; k <= kmax; ++k) {
            float ljk = rowj[k];
            float* rowjk = B + (size_t)(j + k) * BAND_STR;
            for (int d = t; d <= kmax - k; d += 128)
                rowjk[d] -= rowj[k + d] * ljk;
        }
        __syncthreads();
    }
    if (t == 0) {
        for (int j = 0; j < GSZ; ++j) {           // L y = Atb
            float s = rb[j];
            int dmax = min(BAND_P, j);
            for (int d = 1; d <= dmax; ++d)
                s -= B[(size_t)(j - d) * BAND_STR + d] * rb[j - d];
            rb[j] = s / B[(size_t)j * BAND_STR];
        }
        for (int j = GSZ - 1; j >= 0; --j) {      // L^T x = y
            float s = rb[j];
            int dmax = min(BAND_P, GSZ - 1 - j);
            for (int d = 1; d <= dmax; ++d)
                s -= B[(size_t)j * BAND_STR + d] * rb[j + d];
            float xv = s / B[(size_t)j * BAND_STR];
            rb[j] = xv;
            out[(b << 10) + j] = xv;
        }
    }
}

// ---------------------------------------------------------------------------
// Host orchestration
// ---------------------------------------------------------------------------
extern "C" void kernel_launch(void* const* d_in, const int* in_sizes, int n_in,
                              void* d_out, int out_size, void* d_ws, size_t ws_size,
                              hipStream_t stream) {
    (void)in_sizes; (void)n_in; (void)out_size; (void)ws_size;
    const float* u   = (const float*)d_in[0];
    const float* rhs = (const float*)d_in[1];
    const float* cw[7]; const float* cb[7];
    for (int i = 0; i < 7; ++i) { cw[i] = (const float*)d_in[2 + 2 * i];
                                  cb[i] = (const float*)d_in[3 + 2 * i]; }
    const float* mw[4]; const float* mb[4];
    for (int i = 0; i < 4; ++i) { mw[i] = (const float*)d_in[16 + 2 * i];
                                  mb[i] = (const float*)d_in[17 + 2 * i]; }
    const float* pin = (const float*)d_in[24];
    const float* s0l = (const float*)d_in[25];
    const float* s1l = (const float*)d_in[26];

    char* ws = (char*)d_ws;
    size_t off = 0;
    auto alloc = [&](size_t bytes) {
        size_t o = off; off += (bytes + 255) & ~(size_t)255; return o;
    };

    static const int Oarr[7]    = { 64, 128, 256, 256, 128,  64, 100};
    static const int Carr[7]    = {100,  64, 128, 256, 256, 128,  64};
    static const int Cparr[7]   = {128,  64, 128, 256, 256, 128,  64};
    static const int Opadarr[7] = { 64, 128, 256, 256, 128,  64, 128};

    uint16_t* X0 = (uint16_t*)(ws + alloc((size_t)BS * LSEQ * 256 * 2));
    uint16_t* X1 = (uint16_t*)(ws + alloc((size_t)BS * LSEQ * 256 * 2));
    uint16_t* Wp[7];
    for (int i = 0; i < 7; ++i)
        Wp[i] = (uint16_t*)(ws + alloc((size_t)5 * Opadarr[i] * Cparr[i] * 2));
    float* Ylast  = (float*)(ws + alloc((size_t)BS * LSEQ * 128 * 4));
    float* pbuf   = (float*)(ws + alloc((size_t)BS * GSZ * 4));
    float* qbuf   = (float*)(ws + alloc((size_t)BS * GSZ * 4));
    float* params = (float*)(ws + alloc(64));
    float* invbuf = (float*)(ws + alloc(64 * 4));
    float* band   = (float*)(ws + alloc((size_t)BS * GSZ * BAND_STR * 4));
    float* Atb    = (float*)(ws + alloc((size_t)BS * GSZ * 4));

    // pack conv weights (bf16, tap-major, zero-padded)
    for (int i = 0; i < 7; ++i) {
        dim3 g(Opadarr[i], 5);
        k_pack_w<<<g, Cparr[i], 0, stream>>>(cw[i], Wp[i], Oarr[i], Carr[i],
                                             Opadarr[i], Cparr[i]);
    }
    // channel-last bf16 input
    { dim3 g(LSEQ, BS); k_transpose_in<<<g, 128, 0, stream>>>(u, X0); }

    // conv chain (WMMA GEMMs); grid = (32768/64 position tiles, OPAD/64)
    k_conv_wmma<128,  64, false, true ><<<dim3(512, 1), 128, 0, stream>>>(X0, Wp[0], cb[0],  64, X1);
    k_conv_wmma< 64, 128, false, true ><<<dim3(512, 2), 128, 0, stream>>>(X1, Wp[1], cb[1], 128, X0);
    k_conv_wmma<128, 256, false, true ><<<dim3(512, 4), 128, 0, stream>>>(X0, Wp[2], cb[2], 256, X1);
    k_conv_wmma<256, 256, false, true ><<<dim3(512, 4), 128, 0, stream>>>(X1, Wp[3], cb[3], 256, X0);
    k_conv_wmma<256, 128, false, true ><<<dim3(512, 2), 128, 0, stream>>>(X0, Wp[4], cb[4], 128, X1);
    k_conv_wmma<128,  64, false, true ><<<dim3(512, 1), 128, 0, stream>>>(X1, Wp[5], cb[5],  64, X0);
    k_conv_wmma< 64, 128, true,  false><<<dim3(512, 2), 128, 0, stream>>>(X0, Wp[6], cb[6], 100, Ylast);

    k_mlp<<<1, 1024, 0, stream>>>(pin, mw[0], mb[0], mw[1], mb[1],
                                  mw[2], mb[2], mw[3], mb[3], params);
    k_steps<<<1, 64, 0, stream>>>(s0l, s1l, invbuf);
    k_upq<<<BS, 1024, 0, stream>>>(Ylast, params, pbuf, qbuf);
    {
        int total = BS * GSZ * BAND_STR;
        k_init_band<<<(total + 255) / 256, 256, 0, stream>>>(band, Atb);
    }
    k_scatter<<<(BS * GSZ) / 256, 256, 0, stream>>>(pbuf, qbuf, invbuf, rhs,
                                                    band, Atb);
    k_cholsolve<<<BS, 128, 0, stream>>>(band, Atb, (float*)d_out);
}
